// MultiHeadAttention_90254442758261
// MI455X (gfx1250) — matrix operand order repaired and checked by reading, not compiled, not run on hardware
//
#include <hip/hip_runtime.h>
#include <hip/hip_bf16.h>

// ---------------------------------------------------------------------------
// CDNA5 (gfx1250) fused causal multi-head attention, bf16 WMMA everywhere.
//   D_MODEL=1024, H=16, d_k=64, B=4, S=2048.
// Pipeline:
//   0) f32 -> bf16 streaming converts (activations + weights)  [BW-bound]
//   1-3) bf16 WMMA GEMMs for Q (x0.125), K, V^T projections
//   4) flash attention (one wave per 16-row q-tile, 32-key blocks;
//      mask applied only on the diagonal block)
//   5) bf16 WMMA GEMM for output projection -> f32
// ---------------------------------------------------------------------------

typedef __attribute__((ext_vector_type(16))) __bf16 v16bf;
typedef __attribute__((ext_vector_type(8)))  float  v8f;

#define DMODEL 1024
#define NHEADS 16
#define DKH    64
#define SEQ    2048
#define BATCH  4

// Native f32->bf16 (backend selects packed bf16 cvt on gfx1250).
__device__ __forceinline__ __bf16 f2bf(float x) { return (__bf16)x; }

union AF { v16bf v; __bf16 b[16]; uint4 q[2]; };

__device__ __forceinline__ v8f zero8() {
  v8f z;
#pragma unroll
  for (int i = 0; i < 8; ++i) z[i] = 0.f;
  return z;
}

__device__ __forceinline__ void pack8(__bf16* d, float4 a, float4 b) {
  d[0] = f2bf(a.x); d[1] = f2bf(a.y); d[2] = f2bf(a.z); d[3] = f2bf(a.w);
  d[4] = f2bf(b.x); d[5] = f2bf(b.y); d[6] = f2bf(b.z); d[7] = f2bf(b.w);
}

__device__ __forceinline__ v8f wmma_bf16(const AF& a, const AF& b, v8f c) {
  // D(16x16,f32) = A(16x32,bf16) x B(32x16,bf16) + C
  return __builtin_amdgcn_wmma_f32_16x16x32_bf16(
      /*neg_a=*/false, a.v, /*neg_b=*/false, b.v,
      /*c_mod=*/(short)0, c, /*reuse_a=*/false, /*reuse_b=*/false);
}

// ---------------------------------------------------------------------------
// Streaming f32 -> bf16 convert, 8 elements (one b128 store) per lane.
// ---------------------------------------------------------------------------
__global__ __launch_bounds__(256) void cvt_f32_bf16(const float* __restrict__ x,
                                                    __bf16* __restrict__ y,
                                                    int n8) {
  const int i = blockIdx.x * blockDim.x + threadIdx.x;
  if (i < n8) {
    const float4 a = ((const float4*)x)[2 * i];
    const float4 b = ((const float4*)x)[2 * i + 1];
    union { __bf16 e[8]; uint4 q; } o;
    pack8(o.e, a, b);
    ((uint4*)y)[i] = o.q;
  }
}

// ---------------------------------------------------------------------------
// GEMM: Y = Xbf @ Wbf^T + bias, optionally scaled, with output re-layout.
//   OMODE : 0 = f32 flat [M, DMODEL]
//           1 = bf16 head-major [B,H,S,d_k]             (Q, K)
//           2 = bf16 head-major transposed [B,H,d_k,S]  (V^T)
// Block = 256 threads = 8 waves; wave tile 32x64 (2x4 WMMA frags);
// block tile 64x256.  grid = (M/64, DMODEL/256).
// Inner loop is pure global_load_b128 + 8 WMMA: no conversion VALU.
// ---------------------------------------------------------------------------
template <int OMODE>
__global__ __launch_bounds__(256) void gemm_wmma(
    const __bf16* __restrict__ X, const __bf16* __restrict__ W,
    const float* __restrict__ bias, void* __restrict__ Yv, float scale) {
  const int lane = threadIdx.x & 31;
  const int wave = threadIdx.x >> 5;
  const int l  = lane & 15;   // column / row-within-tile index
  const int hh = lane >> 4;   // lane half (splits K)
  const int wm = wave & 1, wn = wave >> 1;
  const int row0 = blockIdx.x * 64 + wm * 32;
  const int col0 = blockIdx.y * 256 + wn * 64;

  v8f acc[2][4];
#pragma unroll
  for (int mi = 0; mi < 2; ++mi)
#pragma unroll
    for (int ni = 0; ni < 4; ++ni) acc[mi][ni] = zero8();

  const __bf16* x0 = X + (size_t)(row0 + l) * DMODEL;
  const __bf16* x1 = X + (size_t)(row0 + 16 + l) * DMODEL;

  for (int k0 = 0; k0 < DMODEL; k0 += 32) {
    AF a[2];
    // A(16x32): lanes 0-15 rows, lane half splits K into two 8-wide chunks.
    a[0].q[0] = *(const uint4*)(x0 + k0 + hh * 8);
    a[0].q[1] = *(const uint4*)(x0 + k0 + 16 + hh * 8);
    a[1].q[0] = *(const uint4*)(x1 + k0 + hh * 8);
    a[1].q[1] = *(const uint4*)(x1 + k0 + 16 + hh * 8);
#pragma unroll
    for (int ni = 0; ni < 4; ++ni) {
      // B(k,n) = W[n][k]: lane = column n, lane half splits K into two 8-wide chunks.
      const int n = col0 + ni * 16 + l;
      const __bf16* wr = W + (size_t)n * DMODEL + k0 + hh * 8;
      AF bf;
      bf.q[0] = *(const uint4*)(wr);
      bf.q[1] = *(const uint4*)(wr + 16);
      acc[0][ni] = wmma_bf16(a[0], bf, acc[0][ni]);
      acc[1][ni] = wmma_bf16(a[1], bf, acc[1][ni]);
    }
  }

  // Epilogue: bias, scale, layout transform.
#pragma unroll
  for (int ni = 0; ni < 4; ++ni) {
    const int c = col0 + ni * 16 + l;
    const float bi = bias[c];
#pragma unroll
    for (int mi = 0; mi < 2; ++mi) {
#pragma unroll
      for (int v = 0; v < 8; ++v) {
        const int r = row0 + mi * 16 + v + 8 * hh;   // C-tile row = vgpr + 8*half
        const float val = (acc[mi][ni][v] + bi) * scale;
        if (OMODE == 0) {
          ((float*)Yv)[(size_t)r * DMODEL + c] = val;
        } else {
          const int bb = r >> 11, ss = r & (SEQ - 1);
          const int hd = c >> 6, dd = c & (DKH - 1);
          __bf16* Y = (__bf16*)Yv;
          if (OMODE == 1)
            Y[(((size_t)bb * NHEADS + hd) * SEQ + ss) * DKH + dd] = f2bf(val);
          else
            Y[(((size_t)bb * NHEADS + hd) * DKH + dd) * SEQ + ss] = f2bf(val);
        }
      }
    }
  }
}

// ---------------------------------------------------------------------------
// One 32-key flash-attention block.  MASKED=true only for the diagonal block;
// interior blocks skip all causal compares/selects.
// ---------------------------------------------------------------------------
template <bool MASKED>
__device__ __forceinline__ void attn_block(
    int kb, int q0, int l, int hh,
    const __bf16* __restrict__ Kb, const __bf16* __restrict__ Vb,
    const AF (&qa)[2], __bf16* __restrict__ ldsrow,
    float (&rmax)[8], float (&rsum)[8], v8f (&ao)[4]) {
  float sfr[2][8];
#pragma unroll
  for (int jt = 0; jt < 2; ++jt) {
    const int kt = kb * 32 + jt * 16;
    // K as B-operand: lane = key column, contiguous d per lane.
    const __bf16* kr = Kb + (size_t)(kt + l) * DKH;
    AF b0, b1;
    b0.q[0] = *(const uint4*)(kr + hh * 16);
    b0.q[1] = *(const uint4*)(kr + hh * 16 + 8);
    b1.q[0] = *(const uint4*)(kr + 32 + hh * 16);
    b1.q[1] = *(const uint4*)(kr + 32 + hh * 16 + 8);
    v8f s = zero8();
    s = wmma_bf16(qa[0], b0, s);
    s = wmma_bf16(qa[1], b1, s);
#pragma unroll
    for (int v = 0; v < 8; ++v) {
      if (MASKED)
        sfr[jt][v] = (kt + l <= q0 + v + 8 * hh) ? s[v] : -__builtin_inff();
      else
        sfr[jt][v] = s[v];
    }
  }
  // Row max across the 16 lanes holding one row (shfl_xor stays in the half).
  float corr[8];
#pragma unroll
  for (int v = 0; v < 8; ++v) {
    float m = fmaxf(sfr[0][v], sfr[1][v]);
    m = fmaxf(m, __shfl_xor(m, 1, 32));
    m = fmaxf(m, __shfl_xor(m, 2, 32));
    m = fmaxf(m, __shfl_xor(m, 4, 32));
    m = fmaxf(m, __shfl_xor(m, 8, 32));
    const float mn = fmaxf(rmax[v], m);
    corr[v] = __expf(rmax[v] - mn);
    rmax[v] = mn;
  }
#pragma unroll
  for (int nt = 0; nt < 4; ++nt)
#pragma unroll
    for (int v = 0; v < 8; ++v) ao[nt][v] *= corr[v];

  // p = exp(s - m); stage to LDS in row-major for A-operand reload.
  float ps[8];
#pragma unroll
  for (int v = 0; v < 8; ++v) ps[v] = 0.f;
#pragma unroll
  for (int jt = 0; jt < 2; ++jt)
#pragma unroll
    for (int v = 0; v < 8; ++v) {
      const float p = __expf(sfr[jt][v] - rmax[v]);   // exp(-inf)=0 under mask
      ps[v] += p;
      ldsrow[(v + 8 * hh) * 32 + jt * 16 + l] = f2bf(p);
    }
#pragma unroll
  for (int v = 0; v < 8; ++v) {
    float s = ps[v];
    s += __shfl_xor(s, 1, 32);
    s += __shfl_xor(s, 2, 32);
    s += __shfl_xor(s, 4, 32);
    s += __shfl_xor(s, 8, 32);
    rsum[v] = rsum[v] * corr[v] + s;
  }

  // CDNA5 split counter: wave's DS stores complete before cross-lane reads.
  asm volatile("s_wait_dscnt 0" ::: "memory");

  AF pa;   // P as A-operand: 16 q-rows x 32 keys
  pa.q[0] = *(const uint4*)&ldsrow[l * 32 + hh * 16];
  pa.q[1] = *(const uint4*)&ldsrow[l * 32 + hh * 16 + 8];
#pragma unroll
  for (int nt = 0; nt < 4; ++nt) {
    // V^T as B-operand: lane = d column, contiguous keys per lane.
    const __bf16* vr = Vb + (size_t)(nt * 16 + l) * SEQ + kb * 32 + hh * 16;
    AF vb;
    vb.q[0] = *(const uint4*)(vr);
    vb.q[1] = *(const uint4*)(vr + 8);
    ao[nt] = wmma_bf16(pa, vb, ao[nt]);
  }
}

// ---------------------------------------------------------------------------
// Flash attention: one wave owns one (b, h, 16-row q-tile); streams 32-key
// blocks with online softmax.  Q pre-scaled by 1/sqrt(d_k).
// ---------------------------------------------------------------------------
__global__ __launch_bounds__(256) void attn_wmma(
    const __bf16* __restrict__ Q, const __bf16* __restrict__ K,
    const __bf16* __restrict__ Vt, __bf16* __restrict__ ctx) {
  __shared__ __bf16 ldsP[8][16 * 32];   // per-wave P staging (C-layout -> A-layout)

  const int lane = threadIdx.x & 31;
  const int wave = threadIdx.x >> 5;
  const int l  = lane & 15;
  const int hh = lane >> 4;
  const int wgid = blockIdx.x * 8 + wave;       // 8192 waves total
  const int bh = wgid >> 7;                     // (b*16 + h)
  const int q0 = (wgid & 127) * 16;
  const int bb = bh >> 4, hd = bh & 15;

  const __bf16* Qb = Q + (size_t)bh * SEQ * DKH;
  const __bf16* Kb = K + (size_t)bh * SEQ * DKH;
  const __bf16* Vb = Vt + (size_t)bh * DKH * SEQ;
  __bf16* ldsrow = &ldsP[wave][0];

  // Q as A-operand: rows q0..q0+15, two k=32 fragments covering d=0..63.
  AF qa[2];
  {
    const __bf16* qr = Qb + (size_t)(q0 + l) * DKH;
#pragma unroll
    for (int kk = 0; kk < 2; ++kk) {
      qa[kk].q[0] = *(const uint4*)(qr + kk * 32 + hh * 16);
      qa[kk].q[1] = *(const uint4*)(qr + kk * 32 + hh * 16 + 8);
    }
  }

  float rmax[8], rsum[8];
#pragma unroll
  for (int v = 0; v < 8; ++v) { rmax[v] = -__builtin_inff(); rsum[v] = 0.f; }
  v8f ao[4];
#pragma unroll
  for (int nt = 0; nt < 4; ++nt) ao[nt] = zero8();

  // Interior blocks (strictly below diagonal): no causal mask needed.
  const int diag = q0 >> 5;
  for (int kb = 0; kb < diag; ++kb)
    attn_block<false>(kb, q0, l, hh, Kb, Vb, qa, ldsrow, rmax, rsum, ao);
  // Diagonal block: apply causal mask.
  attn_block<true>(diag, q0, l, hh, Kb, Vb, qa, ldsrow, rmax, rsum, ao);

  // Normalize and write context back in [B, S, H*d_k] = [B, S, DMODEL] bf16.
#pragma unroll
  for (int nt = 0; nt < 4; ++nt) {
#pragma unroll
    for (int v = 0; v < 8; ++v) {
      const int ss = q0 + v + 8 * hh;
      const int cc = hd * DKH + nt * 16 + l;
      ctx[((size_t)bb * SEQ + ss) * DMODEL + cc] = f2bf(ao[nt][v] / rsum[v]);
    }
  }
}

// ---------------------------------------------------------------------------
// Host-side launch.
// Workspace (bf16 elems): Q|K|V^T|ctx (4 x 8.4M) + qbf|kbf|vbf (3 x 8.4M)
//                         + Wq|Wk|Wv|Wo (4 x 1.05M)  ~= 126 MB.
// ---------------------------------------------------------------------------
extern "C" void kernel_launch(void* const* d_in, const int* in_sizes, int n_in,
                              void* d_out, int out_size, void* d_ws, size_t ws_size,
                              hipStream_t stream) {
  const float* q  = (const float*)d_in[0];
  const float* k  = (const float*)d_in[1];
  const float* v  = (const float*)d_in[2];
  // d_in[3] is the causal mask; implemented analytically in attn_wmma.
  const float* Wq = (const float*)d_in[4];
  const float* bq = (const float*)d_in[5];
  const float* Wk = (const float*)d_in[6];
  const float* bk = (const float*)d_in[7];
  const float* Wv = (const float*)d_in[8];
  const float* bv = (const float*)d_in[9];
  const float* Wo = (const float*)d_in[10];
  const float* bo = (const float*)d_in[11];

  const size_t actN = (size_t)BATCH * SEQ * DMODEL;   // 8,388,608
  const size_t wN   = (size_t)DMODEL * DMODEL;        // 1,048,576

  __bf16* wsQ = (__bf16*)d_ws;      // head-major Q (pre-scaled)
  __bf16* wsK = wsQ + actN;         // head-major K
  __bf16* wsV = wsK + actN;         // head-major V^T
  __bf16* wsC = wsV + actN;         // attention context
  __bf16* qbf = wsC + actN;         // bf16 activations
  __bf16* kbf = qbf + actN;
  __bf16* vbf = kbf + actN;
  __bf16* wqb = vbf + actN;         // bf16 weights
  __bf16* wkb = wqb + wN;
  __bf16* wvb = wkb + wN;
  __bf16* wob = wvb + wN;

  // 0) streaming converts (8 elems per lane)
  {
    const int na8 = (int)(actN / 8), nw8 = (int)(wN / 8);
    cvt_f32_bf16<<<na8 / 256, 256, 0, stream>>>(q, qbf, na8);
    cvt_f32_bf16<<<na8 / 256, 256, 0, stream>>>(k, kbf, na8);
    cvt_f32_bf16<<<na8 / 256, 256, 0, stream>>>(v, vbf, na8);
    cvt_f32_bf16<<<nw8 / 256, 256, 0, stream>>>(Wq, wqb, nw8);
    cvt_f32_bf16<<<nw8 / 256, 256, 0, stream>>>(Wk, wkb, nw8);
    cvt_f32_bf16<<<nw8 / 256, 256, 0, stream>>>(Wv, wvb, nw8);
    cvt_f32_bf16<<<nw8 / 256, 256, 0, stream>>>(Wo, wob, nw8);
  }

  const dim3 gg(BATCH * SEQ / 64, DMODEL / 256);  // (128, 4)
  const float inv_sqrt_dk = 0.125f;               // 1/sqrt(64), folded into Q

  gemm_wmma<1><<<gg, 256, 0, stream>>>(qbf, wqb, bq, wsQ, inv_sqrt_dk);
  gemm_wmma<1><<<gg, 256, 0, stream>>>(kbf, wkb, bk, wsK, 1.0f);
  gemm_wmma<2><<<gg, 256, 0, stream>>>(vbf, wvb, bv, wsV, 1.0f);

  attn_wmma<<<BATCH * NHEADS * (SEQ / 16) / 8, 256, 0, stream>>>(wsQ, wsK, wsV, wsC);

  gemm_wmma<0><<<gg, 256, 0, stream>>>(wsC, wob, bo, d_out, 1.0f);
}
